// WindowAttention_63213328662605
// MI455X (gfx1250) — matrix operand order repaired and checked by reading, not compiled, not run on hardware
//
#include <hip/hip_runtime.h>

// ---------------------------------------------------------------------------
// Fused Swin window attention for gfx1250 (MI455X).
//   B=32, H=W=64, C=384, window 8x8 (T=64 tokens), NH=12 heads, HD=32.
//   One workgroup (128 threads = 4 wave32) per window; 3 heads per wave.
//   All GEMMs via v_wmma_f32_16x16x32_bf16 (f32 accumulate).
// ---------------------------------------------------------------------------

#define C_     384
#define T_     64
#define NH_    12
#define HD_    32
#define SCALE_ 0.17677669529663687f   // 32^-0.5
#define NWIN_  2048                   // 32 * 8 * 8

typedef __bf16 bf16;
typedef __attribute__((ext_vector_type(16))) __bf16 v16bf;
typedef __attribute__((ext_vector_type(8)))  __bf16 v8bf;
typedef __attribute__((ext_vector_type(8)))  float  v8f;
typedef __attribute__((ext_vector_type(4)))  float  f32x4;

#define WMMA_BF16(a, b, c) \
  __builtin_amdgcn_wmma_f32_16x16x32_bf16(false, (a), false, (b), (short)0, (c), false, false)

#define LXS 392   // LDS row stride for X / O staging (bf16): 784B rows, 16B aligned

// ---- A fragment (16x32 bf16, row-striped within lane; ISA 7.12.2) ----------
// lanes 0-15: row M=lane,   K chunks {k0..k0+7, k0+16..k0+23}
// lanes 16-31: row M=lane-16, chunks shifted by +8
__device__ __forceinline__ v16bf load_a16(const bf16* base, int stride, int m0, int k0) {
  const int lane = threadIdx.x & 31;
  const int m = m0 + (lane & 15);
  const int k = k0 + ((lane >> 4) << 3);
  const bf16* p = base + m * stride + k;
  v8bf lo = *(const v8bf*)(p);        // K = k .. k+7      (ds/global _b128)
  v8bf hi = *(const v8bf*)(p + 16);   // K = k+16 .. k+23
  v16bf r;
#pragma unroll
  for (int i = 0; i < 8; ++i) { r[i] = lo[i]; r[i + 8] = hi[i]; }
  return r;
}

// ---- B fragment (32x16 bf16, col per lane; source stored transposed -------
//      baseT[n][k], k contiguous). lane n=l%16, K chunks {k0..k0+7, k0+16..k0+23} (+8 for l>=16).
__device__ __forceinline__ v16bf load_b16(const bf16* baseT, int stride, int n0, int k0) {
  const int lane = threadIdx.x & 31;
  const int n = n0 + (lane & 15);
  const int k = k0 + ((lane >> 4) << 3);
  const bf16* p = baseT + n * stride + k;
  v8bf lo = *(const v8bf*)(p);        // K = k .. k+7
  v8bf hi = *(const v8bf*)(p + 16);   // K = k+16 .. k+23
  v16bf r;
#pragma unroll
  for (int i = 0; i < 8; ++i) { r[i] = lo[i]; r[i + 8] = hi[i]; }
  return r;
}

// ---- 64x32 = lX(64x384) @ wT[colbase..colbase+32]  (+bias)*scale -----------
// dst: row-major [64][dstride] if !TRANS, else transposed [32][dstride].
template <bool TRANS>
__device__ __forceinline__ void gemm_qkv(const bf16* lX, const bf16* wT, const float* bias,
                                         int colbase, float scale, bf16* dst, int dstride) {
  v8f acc[4][2];
#pragma unroll
  for (int mt = 0; mt < 4; ++mt)
#pragma unroll
    for (int nt = 0; nt < 2; ++nt) acc[mt][nt] = (v8f){};

#pragma unroll 2
  for (int ks = 0; ks < 12; ++ks) {
    const int k0 = ks * 32;
    v16bf a0 = load_a16(lX, LXS,  0, k0);
    v16bf a1 = load_a16(lX, LXS, 16, k0);
    v16bf a2 = load_a16(lX, LXS, 32, k0);
    v16bf a3 = load_a16(lX, LXS, 48, k0);
    v16bf b0 = load_b16(wT, C_, colbase +  0, k0);
    v16bf b1 = load_b16(wT, C_, colbase + 16, k0);
    acc[0][0] = WMMA_BF16(a0, b0, acc[0][0]);
    acc[1][0] = WMMA_BF16(a1, b0, acc[1][0]);
    acc[2][0] = WMMA_BF16(a2, b0, acc[2][0]);
    acc[3][0] = WMMA_BF16(a3, b0, acc[3][0]);
    acc[0][1] = WMMA_BF16(a0, b1, acc[0][1]);
    acc[1][1] = WMMA_BF16(a1, b1, acc[1][1]);
    acc[2][1] = WMMA_BF16(a2, b1, acc[2][1]);
    acc[3][1] = WMMA_BF16(a3, b1, acc[3][1]);
  }

  const int lane  = threadIdx.x & 31;
  const int ncol  = lane & 15;
  const int half8 = (lane >> 4) << 3;   // D layout: rows m0+j (+8 for hi lanes)
#pragma unroll
  for (int nt = 0; nt < 2; ++nt) {
    const float bi = bias[colbase + nt * 16 + ncol];
#pragma unroll
    for (int mt = 0; mt < 4; ++mt)
#pragma unroll
      for (int j = 0; j < 8; ++j) {
        const float v = (acc[mt][nt][j] + bi) * scale;
        const int m = mt * 16 + j + half8;
        const int n = nt * 16 + ncol;
        if (TRANS) dst[n * dstride + m] = (bf16)v;
        else       dst[m * dstride + n] = (bf16)v;
      }
  }
}

// ---------------------------------------------------------------------------
__global__ void cvt_weights(const float* __restrict__ w_qkv, const float* __restrict__ w_proj,
                            bf16* __restrict__ wqkvT, bf16* __restrict__ wprojT) {
  const int i = blockIdx.x * blockDim.x + threadIdx.x;
  if (i < C_ * 3 * C_) {                       // w_qkv [384][1152] -> wqkvT [1152][384]
    const int r = i / (3 * C_), c = i % (3 * C_);
    wqkvT[c * C_ + r] = (bf16)w_qkv[i];
  } else if (i < C_ * 3 * C_ + C_ * C_) {      // w_proj [384][384] -> wprojT [384][384]
    const int j = i - C_ * 3 * C_;
    const int r = j / C_, c = j % C_;
    wprojT[c * C_ + r] = (bf16)w_proj[j];
  }
}

// ---------------------------------------------------------------------------
__global__ __launch_bounds__(128, 1)
void win_attn(const float* __restrict__ x, const float* __restrict__ b_qkv,
              const float* __restrict__ b_proj, const bf16* __restrict__ wqkvT,
              const bf16* __restrict__ wprojT, float* __restrict__ out) {
  __shared__ bf16 lX[64 * LXS];        // window activations, bf16
  __shared__ bf16 lO[64 * LXS];        // concat attention output (64 x 384)
  __shared__ bf16 scr[4 * 6144];       // per-wave scratch: Q | K | VT (P overlays Q+K)

  const int wid  = blockIdx.x;
  const int bb   = wid >> 6;
  const int wy   = (wid >> 3) & 7;
  const int wx   = wid & 7;
  const int tid  = threadIdx.x;
  const int wave = tid >> 5;
  const int lane = tid & 31;
  const int ncol = lane & 15;
  const int half8 = (lane >> 4) << 3;

  // ---- Phase A: stage window into LDS (f32 -> bf16), 64 tokens x 384 ch ----
  for (int idx = tid; idx < 64 * 96; idx += 128) {
    const int t  = idx / 96;
    const int c4 = (idx % 96) * 4;
    const int py = wy * 8 + (t >> 3);
    const int px = wx * 8 + (t & 7);
    const f32x4 v = *(const f32x4*)(x + ((((size_t)bb * 64 + py) * 64 + px) * C_ + c4));
    bf16* q = lX + t * LXS + c4;
    q[0] = (bf16)v[0]; q[1] = (bf16)v[1]; q[2] = (bf16)v[2]; q[3] = (bf16)v[3];
  }
  __syncthreads();

  // ---- Phase B: each wave processes 3 heads end-to-end ---------------------
  bf16* scrQ  = scr + wave * 6144;     // [64][32] row-major
  bf16* scrK  = scrQ + 2048;           // [64][32] row-major
  bf16* scrVT = scrQ + 4096;           // [32][64] (V transposed: dim-major)
  bf16* scrP  = scrQ;                  // [64][64] overlays Q+K once frags hoisted

#pragma unroll 1
  for (int hi = 0; hi < 3; ++hi) {
    const int h = wave * 3 + hi;

    // Q (scale folded), K, V^T — three 64x32 GEMMs against L2-resident weights
    gemm_qkv<false>(lX, wqkvT, b_qkv,           h * HD_, SCALE_, scrQ, 32);
    gemm_qkv<false>(lX, wqkvT, b_qkv,      C_ + h * HD_, 1.0f,   scrK, 32);
    gemm_qkv<true >(lX, wqkvT, b_qkv, 2 * C_ + h * HD_, 1.0f,   scrVT, 64);

    // Hoist Q (A-frags) and K (B-frags) so P may overlay their LDS.
    v16bf aq[4], bk[4];
#pragma unroll
    for (int mt = 0; mt < 4; ++mt) aq[mt] = load_a16(scrQ, 32, mt * 16, 0);
#pragma unroll
    for (int nt = 0; nt < 4; ++nt) bk[nt] = load_b16(scrK, 32, nt * 16, 0);

    // S = Q K^T (16x64 row-block at a time) -> softmax -> P (bf16, LDS)
#pragma unroll 1
    for (int mt = 0; mt < 4; ++mt) {
      v8f s[4];
#pragma unroll
      for (int nt = 0; nt < 4; ++nt) s[nt] = (v8f){};
#pragma unroll
      for (int nt = 0; nt < 4; ++nt) s[nt] = WMMA_BF16(aq[mt], bk[nt], s[nt]);

      // Row softmax: a row lives in one accum slot across 16 lanes + 4 tiles.
#pragma unroll
      for (int j = 0; j < 8; ++j) {
        float mx = s[0][j];
#pragma unroll
        for (int nt = 1; nt < 4; ++nt) mx = fmaxf(mx, s[nt][j]);
#pragma unroll
        for (int msk = 8; msk >= 1; msk >>= 1) mx = fmaxf(mx, __shfl_xor(mx, msk, 32));
        float sum = 0.f;
        float e[4];
#pragma unroll
        for (int nt = 0; nt < 4; ++nt) { e[nt] = __expf(s[nt][j] - mx); sum += e[nt]; }
#pragma unroll
        for (int msk = 8; msk >= 1; msk >>= 1) sum += __shfl_xor(sum, msk, 32);
        const float inv = 1.0f / sum;
        const int row = mt * 16 + j + half8;
#pragma unroll
        for (int nt = 0; nt < 4; ++nt)
          scrP[row * 64 + nt * 16 + ncol] = (bf16)(e[nt] * inv);
      }
    }

    // O = P V  (64x32, K=64 over two k-steps), write into lO[:, h*32 ..]
    v8f o[4][2];
#pragma unroll
    for (int mt = 0; mt < 4; ++mt)
#pragma unroll
      for (int nt = 0; nt < 2; ++nt) o[mt][nt] = (v8f){};
#pragma unroll
    for (int kt = 0; kt < 2; ++kt) {
      const int k0 = kt * 32;
      v16bf ap0 = load_a16(scrP, 64,  0, k0);
      v16bf ap1 = load_a16(scrP, 64, 16, k0);
      v16bf ap2 = load_a16(scrP, 64, 32, k0);
      v16bf ap3 = load_a16(scrP, 64, 48, k0);
      v16bf bv0 = load_b16(scrVT, 64,  0, k0);
      v16bf bv1 = load_b16(scrVT, 64, 16, k0);
      o[0][0] = WMMA_BF16(ap0, bv0, o[0][0]);
      o[1][0] = WMMA_BF16(ap1, bv0, o[1][0]);
      o[2][0] = WMMA_BF16(ap2, bv0, o[2][0]);
      o[3][0] = WMMA_BF16(ap3, bv0, o[3][0]);
      o[0][1] = WMMA_BF16(ap0, bv1, o[0][1]);
      o[1][1] = WMMA_BF16(ap1, bv1, o[1][1]);
      o[2][1] = WMMA_BF16(ap2, bv1, o[2][1]);
      o[3][1] = WMMA_BF16(ap3, bv1, o[3][1]);
    }
#pragma unroll
    for (int nt = 0; nt < 2; ++nt)
#pragma unroll
      for (int mt = 0; mt < 4; ++mt)
#pragma unroll
        for (int j = 0; j < 8; ++j) {
          const int m = mt * 16 + j + half8;
          const int n = h * HD_ + nt * 16 + ncol;
          lO[m * LXS + n] = (bf16)o[mt][nt][j];
        }
  }
  __syncthreads();

  // ---- Phase C: projection out = lO(64x384) @ w_proj + b_proj --------------
  // 24 N-tiles split 6 per wave; write f32 straight to window-reversed output.
#pragma unroll 1
  for (int mt = 0; mt < 4; ++mt) {
    v8f acc[6];
#pragma unroll
    for (int nn = 0; nn < 6; ++nn) acc[nn] = (v8f){};
#pragma unroll 1
    for (int ks = 0; ks < 12; ++ks) {
      const int k0 = ks * 32;
      v16bf a = load_a16(lO, LXS, mt * 16, k0);
#pragma unroll
      for (int nn = 0; nn < 6; ++nn) {
        v16bf b = load_b16(wprojT, C_, (wave * 6 + nn) * 16, k0);
        acc[nn] = WMMA_BF16(a, b, acc[nn]);
      }
    }
#pragma unroll
    for (int nn = 0; nn < 6; ++nn) {
      const int n = (wave * 6 + nn) * 16 + ncol;
      const float bi = b_proj[n];
#pragma unroll
      for (int j = 0; j < 8; ++j) {
        const int t  = mt * 16 + j + half8;
        const int py = wy * 8 + (t >> 3);
        const int px = wx * 8 + (t & 7);
        out[(((size_t)bb * 64 + py) * 64 + px) * C_ + n] = acc[nn][j] + bi;
      }
    }
  }
}

// ---------------------------------------------------------------------------
extern "C" void kernel_launch(void* const* d_in, const int* in_sizes, int n_in,
                              void* d_out, int out_size, void* d_ws, size_t ws_size,
                              hipStream_t stream) {
  const float* x      = (const float*)d_in[0];
  const float* w_qkv  = (const float*)d_in[1];
  const float* b_qkv  = (const float*)d_in[2];
  const float* w_proj = (const float*)d_in[3];
  const float* b_proj = (const float*)d_in[4];
  float* out = (float*)d_out;

  bf16* wqkvT  = (bf16*)d_ws;                  // [1152][384] bf16
  bf16* wprojT = wqkvT + 3 * C_ * C_;          // [384][384]  bf16

  const int nconv = C_ * 3 * C_ + C_ * C_;     // 589824
  cvt_weights<<<(nconv + 255) / 256, 256, 0, stream>>>(w_qkv, w_proj, wqkvT, wprojT);
  win_attn<<<NWIN_, 128, 0, stream>>>(x, b_qkv, b_proj, wqkvT, wprojT, out);
}
